// MLA_45999099740784
// MI455X (gfx1250) — matrix operand order repaired and checked by reading, not compiled, not run on hardware
//
#include <hip/hip_runtime.h>
#include <hip/hip_bf16.h>

typedef _Float16 v16h __attribute__((ext_vector_type(16)));
typedef float    v8f  __attribute__((ext_vector_type(8)));

#define DM   3072
#define NH   24
#define HD   128
#define QP   1536
#define KVP  2048
#define RD   64
#define KVD  2112      // KVP + RD
#define KVU  4608      // DM + NH*RD = NH*192
#define BB   2
#define SS   2048
#define NROW 4096      // B*S

// ---------------- fragment loaders (ISA 7.12.2 layouts, wave32) -------------

// A operand: 16x32 f16, row-major source, lda in elements.
// lane<16 : row=lane, halfs[0..7]=K k0..k0+7, halfs[8..15]=K k0+16..k0+23
// lane>=16: row=lane-16, halfs[0..7]=K k0+8..k0+15, halfs[8..15]=K k0+24..k0+31
__device__ inline v16h load_fragA(const _Float16* A, int lda, int row0, int k0, int lane) {
    int r   = lane & 15;
    int hi8 = (lane >= 16) ? 8 : 0;
    const _Float16* p = A + (size_t)(row0 + r) * lda + k0 + hi8;
    union { v16h h; uint4 u[2]; } f;
    f.u[0] = *(const uint4*)(p);
    f.u[1] = *(const uint4*)(p + 16);
    return f.h;
}

// B operand: 32x16 f16 from *transposed* B (Bt is N x K row-major, ldbt elems).
// lane<16 : col=n0+lane, halfs[0..7]=K k0..k0+7, halfs[8..15]=K k0+16..k0+23; lane>=16: col=n0+lane-16, K k0+8..k0+15, k0+24..k0+31
__device__ inline v16h load_fragB(const _Float16* Bt, int ldbt, int n0, int k0, int lane) {
    int c    = lane & 15;
    int hi16 = (lane >= 16) ? 8 : 0;
    const _Float16* p = Bt + (size_t)(n0 + c) * ldbt + k0 + hi16;
    union { v16h h; uint4 u[2]; } f;
    f.u[0] = *(const uint4*)(p);
    f.u[1] = *(const uint4*)(p + 16);
    return f.h;
}

// ---------------- conversion kernels ---------------------------------------

__global__ __launch_bounds__(256) void k_cvt(const float* src, _Float16* dst, int n) {
    int i = blockIdx.x * 256 + threadIdx.x;
    if (i < n) dst[i] = (_Float16)src[i];
}

// dst (N x K) = transpose of src (K x N), fp32 -> f16
__global__ __launch_bounds__(256) void k_cvt_t(const float* src, _Float16* dst, int K, int N) {
    int i = blockIdx.x * 256 + threadIdx.x;
    if (i < K * N) {
        int k = i / N, n = i - k * N;
        dst[(size_t)n * K + k] = (_Float16)src[i];
    }
}

// ---------------- RMSNorm ---------------------------------------------------

__global__ __launch_bounds__(256) void k_rmsnorm(const float* __restrict__ x,
                                                 const float* __restrict__ scale,
                                                 _Float16* __restrict__ xn) {
    __shared__ float red[256];
    int row = blockIdx.x;
    const float* xr = x + (size_t)row * DM;
    float s = 0.f;
    for (int c = threadIdx.x; c < DM; c += 256) { float v = xr[c]; s += v * v; }
    red[threadIdx.x] = s;
    __syncthreads();
    for (int st = 128; st > 0; st >>= 1) {
        if (threadIdx.x < st) red[threadIdx.x] += red[threadIdx.x + st];
        __syncthreads();
    }
    float r = rsqrtf(red[0] * (1.0f / DM) + 1e-6f);
    _Float16* on = xn + (size_t)row * DM;
    for (int c = threadIdx.x; c < DM; c += 256)
        on[c] = (_Float16)(scale[c] * xr[c] * r);
}

// ---------------- generic WMMA GEMM: C = A(MxK) * Bt(NxK)^T -----------------
// block = 128 threads (4 waves); wave computes 32 x 64; grid = (ceil(N/256), M/32)
// K must be a multiple of 64 (true for all uses: 3072/1536/2048).

struct Frags { v16h a0, a1, b[4]; };

__device__ inline Frags load_frags(const _Float16* A, int lda, int m0,
                                   const _Float16* Bt, int ldbt, int n0,
                                   int k0, int lane) {
    Frags f;
    f.a0 = load_fragA(A, lda, m0,      k0, lane);
    f.a1 = load_fragA(A, lda, m0 + 16, k0, lane);
#pragma unroll
    for (int t = 0; t < 4; ++t) f.b[t] = load_fragB(Bt, ldbt, n0 + t * 16, k0, lane);
    return f;
}

__device__ inline void do_wmma(v8f acc[2][4], const Frags& f) {
#pragma unroll
    for (int t = 0; t < 4; ++t) {
        acc[0][t] = __builtin_amdgcn_wmma_f32_16x16x32_f16(false, f.a0, false, f.b[t],
                                                           (short)0, acc[0][t], false, false);
        acc[1][t] = __builtin_amdgcn_wmma_f32_16x16x32_f16(false, f.a1, false, f.b[t],
                                                           (short)0, acc[1][t], false, false);
    }
}

__global__ __launch_bounds__(128) void k_gemm(const _Float16* __restrict__ A, int lda,
                                              const _Float16* __restrict__ Bt, int ldbt,
                                              float* __restrict__ C32,
                                              _Float16* __restrict__ C16,
                                              int ldc, int N, int K) {
    int w = threadIdx.x >> 5, lane = threadIdx.x & 31;
    int m0 = blockIdx.y * 32;
    int n0 = blockIdx.x * 256 + w * 64;
    if (n0 >= N) return;                       // wave-uniform
    v8f acc[2][4] = {};

    // software-pipelined ping-pong over k (step 32, unrolled x2, peeled tail)
    Frags cur = load_frags(A, lda, m0, Bt, ldbt, n0, 0, lane);
    int k0 = 0;
    for (; k0 < K - 64; k0 += 64) {
        // prefetch two k-blocks ahead (global_prefetch_b8)
        int kp = (k0 + 128 < K) ? k0 + 128 : k0;
        __builtin_prefetch(A  + (size_t)(m0 + (lane & 15)) * lda + kp, 0, 1);
        __builtin_prefetch(Bt + (size_t)(n0 + (lane & 15)) * ldbt + kp, 0, 1);
        Frags nxt = load_frags(A, lda, m0, Bt, ldbt, n0, k0 + 32, lane);
        do_wmma(acc, cur);
        cur = load_frags(A, lda, m0, Bt, ldbt, n0, k0 + 64, lane);
        do_wmma(acc, nxt);
    }
    Frags nxt = load_frags(A, lda, m0, Bt, ldbt, n0, K - 32, lane);
    do_wmma(acc, cur);
    do_wmma(acc, nxt);

    int cb = lane & 15;
    int hi8 = (lane >= 16) ? 8 : 0;
#pragma unroll
    for (int mi = 0; mi < 2; ++mi)
#pragma unroll
        for (int t = 0; t < 4; ++t)
#pragma unroll
            for (int v = 0; v < 8; ++v) {
                size_t off = (size_t)(m0 + mi * 16 + v + hi8) * ldc + (n0 + t * 16 + cb);
                float val = acc[mi][t][v];
                if (C32) C32[off] = val;
                if (C16) C16[off] = (_Float16)val;
            }
}

// ---------------- RoPE + Q/K/V assembly ------------------------------------
// grid = (S, B*NH), block = 128 (one thread per head-dim element)
// V is written TRANSPOSED per head: fvt[bh][d][s]  (128 x S) so attention can
// load its WMMA B operand with contiguous b128 loads.

__global__ __launch_bounds__(128) void k_rope(const _Float16* __restrict__ q16,
                                              const _Float16* __restrict__ kvu16,
                                              const float*   __restrict__ kvd32,
                                              _Float16* __restrict__ fq,
                                              _Float16* __restrict__ fk,
                                              _Float16* __restrict__ fvt) {
    int s = blockIdx.x, bh = blockIdx.y;
    int b = bh / NH, h = bh - b * NH;
    int d = threadIdx.x;
    size_t row  = (size_t)b * SS + s;
    const _Float16* qr  = q16  + row * DM  + h * HD;
    const _Float16* kur = kvu16 + row * KVU + h * (HD + RD);
    const float*    krr = kvd32 + row * KVD + KVP;
    size_t orow = ((size_t)bh * SS + s) * HD;

    fvt[((size_t)bh * HD + d) * SS + s] = kur[RD + d];   // value = kv_up[..., 64:192], transposed

    float qv, kv;
    if (d < RD) {
        qv = (float)qr[d];                            // q_no_rope
        kv = (float)kur[d];                           // key (kv_up[..., 0:64])
    } else {
        int j = d - RD;                               // 0..63
        int i = j & 31;
        float ang = (float)s * __expf(-(float)i * 0.14391156831212787f); // ln(1e4)/64
        float cs = cosf(ang), sn = sinf(ang);
        float tq = (float)qr[RD + j];
        float pq = (j < 32) ? (float)qr[RD + j + 32] : (float)qr[RD + j - 32];
        qv = (j < 32) ? tq * cs - pq * sn : tq * cs + pq * sn;
        float tk = krr[j];
        float pk = (j < 32) ? krr[j + 32] : krr[j - 32];
        kv = (j < 32) ? tk * cs - pk * sn : tk * cs + pk * sn;
    }
    fq[orow + d] = (_Float16)qv;
    fk[orow + d] = (_Float16)kv;
}

// ---------------- causal flash attention (WMMA) -----------------------------
// grid = (S/64, B*NH), block = 128; each wave owns 16 query rows, 32-key blocks

__global__ __launch_bounds__(128) void k_attn(const _Float16* __restrict__ fq,
                                              const _Float16* __restrict__ fk,
                                              const _Float16* __restrict__ fvt,
                                              _Float16* __restrict__ ao) {
    __shared__ _Float16 psh[4 * 16 * 32];
    int w = threadIdx.x >> 5, lane = threadIdx.x & 31;
    int bh = blockIdx.y;
    int b = bh / NH, h = bh - b * NH;
    int qbase = blockIdx.x * 64 + w * 16;
    const _Float16* Q  = fq  + (size_t)bh * SS * HD;
    const _Float16* Kp = fk  + (size_t)bh * SS * HD;
    const _Float16* Vt = fvt + (size_t)bh * HD * SS;   // (128 x S) row-major

    v16h qf[4];
#pragma unroll
    for (int c = 0; c < 4; ++c) qf[c] = load_fragA(Q, HD, qbase, c * 32, lane);

    float m_run[8], l_run[8];
    v8f acc[8];
#pragma unroll
    for (int v = 0; v < 8; ++v) { m_run[v] = -__builtin_inff(); l_run[v] = 0.f; }
#pragma unroll
    for (int f = 0; f < 8; ++f) acc[f] = (v8f){};

    const float sc = 0.08838834764831845f;            // 1/sqrt(128)
    int cb  = lane & 15;
    int hi8 = (lane >= 16) ? 8 : 0;
    _Float16* pw = psh + w * 512;
    int nkb = (qbase + 15) / 32 + 1;

    for (int kb = 0; kb < nkb; ++kb) {
        int kbase = kb * 32;
        v8f s0 = (v8f){}, s1 = (v8f){};
#pragma unroll
        for (int c = 0; c < 4; ++c) {
            v16h bk0 = load_fragB(Kp, HD, kbase,      c * 32, lane);
            s0 = __builtin_amdgcn_wmma_f32_16x16x32_f16(false, qf[c], false, bk0, (short)0, s0, false, false);
            v16h bk1 = load_fragB(Kp, HD, kbase + 16, c * 32, lane);
            s1 = __builtin_amdgcn_wmma_f32_16x16x32_f16(false, qf[c], false, bk1, (short)0, s1, false, false);
        }
#pragma unroll
        for (int v = 0; v < 8; ++v) {
            int row = qbase + v + hi8;
            float a0 = s0[v] * sc; if (kbase + cb      > row) a0 = -1e30f;
            float a1 = s1[v] * sc; if (kbase + 16 + cb > row) a1 = -1e30f;
            float mx = fmaxf(a0, a1);
            mx = fmaxf(mx, __shfl_xor(mx, 1, 32));
            mx = fmaxf(mx, __shfl_xor(mx, 2, 32));
            mx = fmaxf(mx, __shfl_xor(mx, 4, 32));
            mx = fmaxf(mx, __shfl_xor(mx, 8, 32));
            float mn    = fmaxf(m_run[v], mx);
            float alpha = __expf(m_run[v] - mn);
            float e0 = __expf(a0 - mn), e1 = __expf(a1 - mn);
            float sm = e0 + e1;
            sm += __shfl_xor(sm, 1, 32);
            sm += __shfl_xor(sm, 2, 32);
            sm += __shfl_xor(sm, 4, 32);
            sm += __shfl_xor(sm, 8, 32);
            l_run[v] = l_run[v] * alpha + sm;
            m_run[v] = mn;
#pragma unroll
            for (int f = 0; f < 8; ++f) acc[f][v] *= alpha;
            pw[(v + hi8) * 32 + cb]      = (_Float16)e0;
            pw[(v + hi8) * 32 + 16 + cb] = (_Float16)e1;
        }
        __builtin_amdgcn_wave_barrier();               // LDS is in-order within a wave
        union { v16h h; uint4 u[2]; } pf;
        const _Float16* pr = psh + w * 512 + cb * 32 + hi8;
        pf.u[0] = *(const uint4*)(pr);
        pf.u[1] = *(const uint4*)(pr + 16);
        __builtin_amdgcn_wave_barrier();
#pragma unroll
        for (int f = 0; f < 8; ++f) {
            v16h bv = load_fragB(Vt, SS, f * 16, kbase, lane);   // contiguous b128 loads
            acc[f] = __builtin_amdgcn_wmma_f32_16x16x32_f16(false, pf.h, false, bv,
                                                            (short)0, acc[f], false, false);
        }
    }

    float invl[8];
#pragma unroll
    for (int v = 0; v < 8; ++v) invl[v] = 1.0f / l_run[v];
#pragma unroll
    for (int f = 0; f < 8; ++f)
#pragma unroll
        for (int v = 0; v < 8; ++v) {
            size_t s_ = (size_t)b * SS + qbase + v + hi8;
            ao[s_ * DM + h * HD + f * 16 + cb] = (_Float16)(acc[f][v] * invl[v]);
        }
}

// ---------------- host driver ----------------------------------------------

extern "C" void kernel_launch(void* const* d_in, const int* in_sizes, int n_in,
                              void* d_out, int out_size, void* d_ws, size_t ws_size,
                              hipStream_t stream) {
    (void)in_sizes; (void)n_in; (void)out_size; (void)ws_size;
    const float* x     = (const float*)d_in[0];
    const float* scale = (const float*)d_in[1];
    const float* w_dq  = (const float*)d_in[2];
    const float* w_uq  = (const float*)d_in[3];
    const float* w_dkv = (const float*)d_in[4];
    const float* w_ukv = (const float*)d_in[5];
    const float* w_o   = (const float*)d_in[6];
    float* out  = (float*)d_out;                       // (B,S,3072)
    float* kvdo = out + (size_t)NROW * DM;             // (B,S,2112) second output

    char* ws = (char*)d_ws;
    size_t off = 0;
    auto alloc = [&](size_t bytes) { char* p = ws + off; off += (bytes + 255) & ~(size_t)255; return p; };
    _Float16* xn16   = (_Float16*)alloc((size_t)NROW * DM  * 2);
    _Float16* wdq16  = (_Float16*)alloc((size_t)QP  * DM   * 2);   // (1536 x 3072) = w_dq^T
    _Float16* wuq16  = (_Float16*)alloc((size_t)DM  * QP   * 2);   // (3072 x 1536) = w_uq^T
    _Float16* wdkv16 = (_Float16*)alloc((size_t)KVD * DM   * 2);   // (2112 x 3072) = w_dkv^T
    _Float16* wukv16 = (_Float16*)alloc((size_t)KVU * KVP  * 2);   // (4608 x 2048) = w_ukv^T
    _Float16* wo16   = (_Float16*)alloc((size_t)DM  * DM   * 2);   // w_o as-is (output @ w_o.T)
    _Float16* qd16   = (_Float16*)alloc((size_t)NROW * QP  * 2);
    _Float16* q16    = (_Float16*)alloc((size_t)NROW * DM  * 2);
    _Float16* kvd16  = (_Float16*)alloc((size_t)NROW * KVD * 2);
    _Float16* kvu16  = (_Float16*)alloc((size_t)NROW * KVU * 2);
    _Float16* fq     = (_Float16*)alloc((size_t)BB * NH * SS * HD * 2);
    _Float16* fk     = (_Float16*)alloc((size_t)BB * NH * SS * HD * 2);
    _Float16* fvt    = (_Float16*)alloc((size_t)BB * NH * HD * SS * 2);  // V transposed per head
    _Float16* ao16   = (_Float16*)alloc((size_t)NROW * DM  * 2);

    auto cdiv = [](int a, int b) { return (a + b - 1) / b; };

    // 1) RMSNorm -> f16
    k_rmsnorm<<<NROW, 256, 0, stream>>>(x, scale, xn16);

    // 2) weight conversions (transposed so B operand is N x K)
    k_cvt_t<<<cdiv(DM * QP,  256), 256, 0, stream>>>(w_dq,  wdq16,  DM,  QP);
    k_cvt_t<<<cdiv(QP * DM,  256), 256, 0, stream>>>(w_uq,  wuq16,  QP,  DM);
    k_cvt_t<<<cdiv(DM * KVD, 256), 256, 0, stream>>>(w_dkv, wdkv16, DM,  KVD);
    k_cvt_t<<<cdiv(KVP * KVU,256), 256, 0, stream>>>(w_ukv, wukv16, KVP, KVU);
    k_cvt  <<<cdiv(DM * DM,  256), 256, 0, stream>>>(w_o,   wo16,   DM * DM);

    // 3) q_down = xn @ w_dq            (4096 x 1536, K=3072)
    k_gemm<<<dim3(cdiv(QP, 256), NROW / 32), 128, 0, stream>>>(
        xn16, DM, wdq16, DM, nullptr, qd16, QP, QP, DM);
    // 4) q = q_down @ w_uq             (4096 x 3072, K=1536)
    k_gemm<<<dim3(cdiv(DM, 256), NROW / 32), 128, 0, stream>>>(
        qd16, QP, wuq16, QP, nullptr, q16, DM, DM, QP);
    // 5) kv_down = xn @ w_dkv          (4096 x 2112, K=3072)  fp32 -> output #2
    k_gemm<<<dim3(cdiv(KVD, 256), NROW / 32), 128, 0, stream>>>(
        xn16, DM, wdkv16, DM, kvdo, kvd16, KVD, KVD, DM);
    // 6) kv_up = kv_down[:, :2048] @ w_ukv   (4096 x 4608, K=2048)
    k_gemm<<<dim3(cdiv(KVU, 256), NROW / 32), 128, 0, stream>>>(
        kvd16, KVD, wukv16, KVP, nullptr, kvu16, KVU, KVU, KVP);

    // 7) RoPE + assemble per-(b,h) Q/K + transposed V
    k_rope<<<dim3(SS, BB * NH), 128, 0, stream>>>(q16, kvu16, kvdo, fq, fk, fvt);

    // 8) causal flash attention
    k_attn<<<dim3(SS / 64, BB * NH), 128, 0, stream>>>(fq, fk, fvt, ao16);

    // 9) output = attn_out @ w_o.T     (4096 x 3072, K=3072) fp32 -> output #1
    k_gemm<<<dim3(cdiv(DM, 256), NROW / 32), 128, 0, stream>>>(
        ao16, DM, wo16, DM, out, nullptr, DM, DM, DM);
}
